// CausalSelfAttention_49718541419115
// MI455X (gfx1250) — matrix operand order repaired and checked by reading, not compiled, not run on hardware
//
#include <hip/hip_runtime.h>
#include <cstdint>
#include <cstddef>

typedef __bf16 bf16;
typedef __attribute__((ext_vector_type(16))) bf16  v16bf;
typedef __attribute__((ext_vector_type(8)))  bf16  v8bf;
typedef __attribute__((ext_vector_type(8)))  float v8f;

static constexpr int B  = 2;
static constexpr int S  = 2048;
static constexpr int D  = 1024;
static constexpr int H  = 16;
static constexpr int HD = 64;          // head dim
static constexpr int D3 = 3 * D;
static constexpr int M  = B * S;       // 4096 rows

// ---------- bf16 <-> f32 (RNE, toolchain-safe bit twiddling) ----------
__device__ __forceinline__ bf16 f2bf(float f) {
  uint32_t u = __builtin_bit_cast(uint32_t, f);
  uint32_t r = u + 0x7fffu + ((u >> 16) & 1u);
  uint16_t h = (uint16_t)(r >> 16);
  return __builtin_bit_cast(bf16, h);
}
__device__ __forceinline__ float bf2f(bf16 b) {
  uint16_t h = __builtin_bit_cast(uint16_t, b);
  uint32_t u = ((uint32_t)h) << 16;
  return __builtin_bit_cast(float, u);
}

// ---------- WMMA helpers ----------
__device__ __forceinline__ v8f wmma_bf16(v16bf a, v16bf b, v8f c) {
  // D = A(16x32 bf16) * B(32x16 bf16) + C(16x16 f32)
  return __builtin_amdgcn_wmma_f32_16x16x32_bf16(false, a, false, b,
                                                 (short)0, c, false, false);
}

// A-fragment (16x32, bf16), row-major source, rowp -> start of this lane's row.
// lane m = l&15, kh = l>>4; elems i -> K = 16*kh + i (matches B-fragment order)
__device__ __forceinline__ v16bf ld_afrag(const bf16* rowp, int koff, int kh) {
  v16bf r;
  *((v8bf*)&r)     = *(const v8bf*)(rowp + koff + 16 * kh);
  *(((v8bf*)&r)+1) = *(const v8bf*)(rowp + koff + 16 * kh + 8);
  return r;
}
// B-fragment (32x16, bf16): lane n = l&15, kh = l>>4; elems i -> K = 16*kh + i
// => 16 contiguous bf16 = one 32B load from column-major (i.e. [N][K]) storage.

// ---------- prep kernels ----------
__global__ void k_cvt(const float* __restrict__ in, bf16* __restrict__ out, int n) {
  int i = blockIdx.x * blockDim.x + threadIdx.x;
  if (i < n) out[i] = f2bf(in[i]);
}

// in[R][C] (f32 row-major) -> out[C][R] (bf16)
__global__ void k_transpose(const float* __restrict__ in, bf16* __restrict__ out,
                            int R, int C) {
  int i = blockIdx.x * blockDim.x + threadIdx.x;
  if (i < R * C) {
    int r = i / C, c = i % C;
    out[(size_t)c * R + r] = f2bf(in[i]);
  }
}

// ---------- generic wave-tiled bf16 GEMM: C[M][N] = A[M][K] * BT[N][K]^T ----------
template <bool F32OUT>
__global__ __launch_bounds__(128) void k_gemm(const bf16* __restrict__ A,
                                              const bf16* __restrict__ BT,
                                              void* __restrict__ outp,
                                              int Mv, int Nv, int Kv) {
  const int lane = threadIdx.x & 31;
  const int n    = lane & 15;
  const int kh   = lane >> 4;
  const int wave = blockIdx.x * (blockDim.x >> 5) + (threadIdx.x >> 5);
  const int ntiles = Nv >> 6;             // 64-wide N tiles
  const int mi = wave / ntiles, ni = wave % ntiles;
  if (mi * 32 >= Mv) return;

  const bf16* arow0 = A + (size_t)(mi * 32 + n) * Kv;
  const bf16* arow1 = arow0 + (size_t)16 * Kv;
  const bf16* bcol  = BT + (size_t)(ni * 64 + n) * Kv;

  const v8f zf = {0.f, 0.f, 0.f, 0.f, 0.f, 0.f, 0.f, 0.f};
  v8f acc[2][4];
#pragma unroll
  for (int a = 0; a < 2; ++a)
#pragma unroll
    for (int f = 0; f < 4; ++f) acc[a][f] = zf;

  for (int k0 = 0; k0 < Kv; k0 += 32) {
    if (k0 + 32 < Kv) {   // prefetch next K-step (global_prefetch_b8)
      __builtin_prefetch((const void*)(arow0 + k0 + 32), 0, 0);
      __builtin_prefetch((const void*)(arow1 + k0 + 32), 0, 0);
      __builtin_prefetch((const void*)(bcol + k0 + 32), 0, 0);
    }
    v16bf a0 = ld_afrag(arow0, k0, kh);
    v16bf a1 = ld_afrag(arow1, k0, kh);
#pragma unroll
    for (int f = 0; f < 4; ++f) {
      v16bf bb = *(const v16bf*)(bcol + (size_t)f * 16 * Kv + k0 + 16 * kh);
      acc[0][f] = wmma_bf16(a0, bb, acc[0][f]);
      acc[1][f] = wmma_bf16(a1, bb, acc[1][f]);
    }
  }

#pragma unroll
  for (int mf = 0; mf < 2; ++mf)
#pragma unroll
    for (int f = 0; f < 4; ++f)
#pragma unroll
      for (int r = 0; r < 8; ++r) {
        const int row = mi * 32 + mf * 16 + r + 8 * kh;
        const int col = ni * 64 + f * 16 + n;
        const float v = acc[mf][f][r];
        if (F32OUT) ((float*)outp)[(size_t)row * Nv + col] = v;
        else        ((bf16*)outp)[(size_t)row * Nv + col]  = f2bf(v);
      }
}

// ---------- RoPE + head layout (Q scaled by 1/sqrt(hd); V stored transposed) ----------
__global__ __launch_bounds__(64) void k_rope(const bf16* __restrict__ qkv,
                                             bf16* __restrict__ qr,
                                             bf16* __restrict__ kr,
                                             bf16* __restrict__ vt) {
  const int j = threadIdx.x;             // 0..63 within head dim
  const int idx = blockIdx.x;            // (b*H + h)*S + s
  const int s = idx & (S - 1);
  const int h = (idx >> 11) & (H - 1);
  const int b = idx >> 15;

  const size_t rowoff = (size_t)(b * S + s) * D3 + h * HD;
  const int jm = j & 31;
  const float inv = __powf(10000.0f, -(float)jm * (1.0f / 32.0f));
  const float ang = (float)s * inv;
  float sn, cs;
  __sincosf(ang, &sn, &cs);
  const float sgn = (j < 32) ? -1.0f : 1.0f;
  const int jp = j ^ 32;

  const float qv = bf2f(qkv[rowoff + j]);
  const float qp = bf2f(qkv[rowoff + jp]);
  const float kv = bf2f(qkv[rowoff + D + j]);
  const float kp = bf2f(qkv[rowoff + D + jp]);
  const float vv = bf2f(qkv[rowoff + 2 * D + j]);

  const size_t ho = ((size_t)(b * H + h) * S + s) * HD + j;
  qr[ho] = f2bf((qv * cs + sgn * qp * sn) * 0.125f);   // fold 1/sqrt(64)
  kr[ho] = f2bf(kv * cs + sgn * kp * sn);
  vt[((size_t)(b * H + h) * HD + j) * S + s] = f2bf(vv);
}

// ---------- flash attention: one wave per 16-row Q tile, 64-key tiles ----------
__global__ __launch_bounds__(32) void k_flash(const bf16* __restrict__ qr,
                                              const bf16* __restrict__ kr,
                                              const bf16* __restrict__ vt,
                                              bf16* __restrict__ attn) {
  __shared__ __align__(16) bf16 pTile[16 * 64];
  const int lane = threadIdx.x & 31;
  const int n    = lane & 15;
  const int kh   = lane >> 4;

  const int idx = blockIdx.x;            // ((b*H + h) * (S/16)) + qt
  const int qt = idx & 127;              // S/16 = 128
  const int h  = (idx >> 7) & 15;
  const int b  = idx >> 11;
  const int qbase = qt * 16;

  const size_t headoff = (size_t)(b * H + h) * S * HD;
  const bf16* qrow  = qr + headoff + (size_t)(qbase + n) * HD;
  const bf16* kbase = kr + headoff;      // [S][HD]
  const bf16* vbase = vt + headoff;      // [HD][S]

  const v16bf qa0 = ld_afrag(qrow, 0, kh);    // hd 0..31
  const v16bf qa1 = ld_afrag(qrow, 32, kh);   // hd 32..63

  const v8f zf = {0.f, 0.f, 0.f, 0.f, 0.f, 0.f, 0.f, 0.f};
  v8f o[4];
#pragma unroll
  for (int f = 0; f < 4; ++f) o[f] = zf;
  float mrow[8], lrow[8];
#pragma unroll
  for (int r = 0; r < 8; ++r) { mrow[r] = -1e30f; lrow[r] = 0.0f; }

  const int qlast = qbase + 15;
  for (int k0 = 0; k0 <= qlast; k0 += 64) {
    // ---- scores for 4 groups of 16 keys: [k0+16j, k0+16j+16) ----
    v8f sc[4];
#pragma unroll
    for (int j = 0; j < 4; ++j) {
      const bf16* kp = kbase + (size_t)(k0 + 16 * j + n) * HD;
      v8f s = zf;
      s = wmma_bf16(qa0, *(const v16bf*)(kp + 16 * kh), s);
      s = wmma_bf16(qa1, *(const v16bf*)(kp + 32 + 16 * kh), s);
      sc[j] = s;
    }
    if (k0 + 64 <= qlast) {  // prefetch next tile's K/V (global_prefetch_b8)
      __builtin_prefetch((const void*)(kbase + (size_t)(k0 + 64 + n) * HD), 0, 0);
      __builtin_prefetch((const void*)(kbase + (size_t)(k0 + 96 + n) * HD), 0, 0);
      __builtin_prefetch((const void*)(vbase + (size_t)n * S + k0 + 64), 0, 0);
    }

    // ---- online softmax over 64 keys at once ----
#pragma unroll
    for (int r = 0; r < 8; ++r) {
      const int rowa = qbase + r + 8 * kh;
      float a[4];
#pragma unroll
      for (int j = 0; j < 4; ++j) {
        const int col = k0 + 16 * j + n;
        a[j] = (col <= rowa) ? sc[j][r] : -1e30f;   // causal mask
      }
      float t = fmaxf(fmaxf(a[0], a[1]), fmaxf(a[2], a[3]));
#pragma unroll
      for (int off = 1; off < 16; off <<= 1) t = fmaxf(t, __shfl_xor(t, off, 32));
      const float mnew  = fmaxf(mrow[r], t);
      const float alpha = __expf(mrow[r] - mnew);
      float p[4];
      float rs = 0.0f;
#pragma unroll
      for (int j = 0; j < 4; ++j) { p[j] = __expf(a[j] - mnew); rs += p[j]; }
#pragma unroll
      for (int off = 1; off < 16; off <<= 1) rs += __shfl_xor(rs, off, 32);
      lrow[r] = lrow[r] * alpha + rs;
      mrow[r] = mnew;
      o[0][r] *= alpha; o[1][r] *= alpha; o[2][r] *= alpha; o[3][r] *= alpha;
      // stash P (C-layout) into LDS as K-major [16][64]
      const int rt = r + 8 * kh;
#pragma unroll
      for (int j = 0; j < 4; ++j) pTile[rt * 64 + 16 * j + n] = f2bf(p[j]);
    }
    __syncthreads();   // single-wave WG: S_NOP, orders LDS write->read
    // ---- P(16x64) @ V(64x64) ----
    const v16bf pa0 = ld_afrag(&pTile[n * 64], 0, kh);
    const v16bf pa1 = ld_afrag(&pTile[n * 64], 32, kh);
#pragma unroll
    for (int f = 0; f < 4; ++f) {
      const bf16* vp = vbase + (size_t)(f * 16 + n) * S + k0;
      o[f] = wmma_bf16(pa0, *(const v16bf*)(vp + 16 * kh), o[f]);
      o[f] = wmma_bf16(pa1, *(const v16bf*)(vp + 32 + 16 * kh), o[f]);
    }
    __syncthreads();
  }

  // normalize and write back as [B, S, H*HD] bf16
#pragma unroll
  for (int f = 0; f < 4; ++f)
#pragma unroll
    for (int r = 0; r < 8; ++r) {
      const int rowa = qbase + r + 8 * kh;
      const float val = o[f][r] / lrow[r];
      attn[(size_t)(b * S + rowa) * D + h * HD + f * 16 + n] = f2bf(val);
    }
}

// ---------- launch ----------
extern "C" void kernel_launch(void* const* d_in, const int* in_sizes, int n_in,
                              void* d_out, int out_size, void* d_ws, size_t ws_size,
                              hipStream_t stream) {
  (void)in_sizes; (void)n_in; (void)out_size; (void)ws_size;
  const float* x     = (const float*)d_in[0];
  const float* w_qkv = (const float*)d_in[1];
  const float* w_out = (const float*)d_in[2];
  float* out = (float*)d_out;

  char* ws = (char*)d_ws;
  size_t off = 0;
  bf16* x16    = (bf16*)(ws + off); off += (size_t)M * D * 2;        // 8 MiB
  bf16* wqkvT  = (bf16*)(ws + off); off += (size_t)D3 * D * 2;       // 6 MiB
  bf16* woutT  = (bf16*)(ws + off); off += (size_t)D * D * 2;        // 2 MiB
  bf16* qkv16  = (bf16*)(ws + off); off += (size_t)M * D3 * 2;       // 24 MiB
  bf16* qrb    = (bf16*)(ws + off); off += (size_t)B * H * S * HD * 2;
  bf16* krb    = (bf16*)(ws + off); off += (size_t)B * H * S * HD * 2;
  bf16* vtb    = (bf16*)(ws + off); off += (size_t)B * H * S * HD * 2;
  bf16* attn16 = (bf16*)(ws + off); off += (size_t)M * D * 2;        // total 72 MiB

  // 1) precision conversion + weight transposes (one-time, L2-resident)
  k_cvt<<<(M * D) / 256, 256, 0, stream>>>(x, x16, M * D);
  k_transpose<<<(D * D3) / 256, 256, 0, stream>>>(w_qkv, wqkvT, D, D3);
  k_transpose<<<(D * D) / 256, 256, 0, stream>>>(w_out, woutT, D, D);

  // 2) QKV GEMM: [4096,1024] x [1024,3072] via bf16 WMMA
  k_gemm<false><<<((M / 32) * (D3 / 64)) / 4, 128, 0, stream>>>(
      x16, wqkvT, (void*)qkv16, M, D3, D);

  // 3) RoPE + head-major layout (+ V transpose)
  k_rope<<<B * H * S, 64, 0, stream>>>(qkv16, qrb, krb, vtb);

  // 4) causal flash attention, all-WMMA inner loops, 64-key tiles
  k_flash<<<B * H * (S / 16), 32, 0, stream>>>(qrb, krb, vtb, attn16);

  // 5) output projection -> f32
  k_gemm<true><<<((M / 32) * (D / 64)) / 4, 128, 0, stream>>>(
      attn16, woutT, (void*)out, M, D, D);
}
